// GCN_55482387530369
// MI455X (gfx1250) — matrix operand order repaired and checked by reading, not compiled, not run on hardware
//
#include <hip/hip_runtime.h>
#include <hip/hip_bf16.h>

#define NNODES 100000
#define NEDGES 1600000
#define FIN    256
#define FHID   64
#define NCLS   40
#define NCP    48   // NCLS padded to 3 WMMA N-tiles

static_assert(NNODES % 16 == 0, "M-tiles must be full or empty");

typedef __attribute__((ext_vector_type(8)))  float  v8f;
typedef __attribute__((ext_vector_type(8)))  __bf16 bf8;
typedef __attribute__((ext_vector_type(16))) __bf16 bf16v;

__device__ __forceinline__ unsigned short f2bf(float f) {
  union { __bf16 b; unsigned short u; } t;
  t.b = (__bf16)f;                       // hardware f32->bf16 (RNE)
  return t.u;
}
__device__ __forceinline__ bf16v cat8(uint4 lo, uint4 hi) {
  union U { uint4 u; bf8 b; };
  U a; a.u = lo; U c; c.u = hi;
  return __builtin_shufflevector(a.b, c.b, 0,1,2,3,4,5,6,7,8,9,10,11,12,13,14,15);
}

// ----------------------------- utility kernels -----------------------------
__global__ void gcn_zero(float* __restrict__ p, int n) {
  int i = blockIdx.x * blockDim.x + threadIdx.x;
  if (i < n) p[i] = 0.0f;
}

__global__ void gcn_degrees(const int* __restrict__ src, const int* __restrict__ dst,
                            float* __restrict__ degOut, float* __restrict__ degIn) {
  int e = blockIdx.x * blockDim.x + threadIdx.x;
  if (e < NEDGES) {
    atomicAdd(&degOut[src[e]], 1.0f);
    atomicAdd(&degIn[dst[e]], 1.0f);
  }
}

__global__ void gcn_rsqrt_inplace(float* __restrict__ p, int n) {
  int i = blockIdx.x * blockDim.x + threadIdx.x;
  if (i < n) p[i] = rsqrtf(fmaxf(p[i], 1.0f));
}

// ------------------------ GEMM1: H1 = (X*rOut) @ W1 ------------------------
// One wave per 16-row M-tile, 4 N-tiles (64 cols). W1^T in LDS as bf16,
// stride 264 halfwords -> consecutive columns land 4 banks apart (no conflicts).
// OOB waves clamp their load row (harmless) and skip the store entirely:
// since NNODES % 16 == 0 a tile is never partial, so EXEC stays all-ones
// through the K-loop and the epilogue needs one uniform branch only.
__global__ void __launch_bounds__(256)
gcn_gemm1(const float* __restrict__ X, const float* __restrict__ W1,
          const float* __restrict__ rOut, float* __restrict__ H1) {
  __shared__ __align__(16) unsigned short sW[FHID * 264];
  const int tid = threadIdx.x;
  for (int idx = tid; idx < FIN * FHID; idx += 256) {
    const int k = idx >> 6, c = idx & 63;          // W1 is [256,64] row-major
    sW[c * 264 + k] = f2bf(W1[idx]);
  }
  __syncthreads();

  const int wave = tid >> 5;
  const int lane = tid & 31;
  const int half = lane >> 4;
  const int l16  = lane & 15;
  const int mbase = (blockIdx.x * 8 + wave) * 16;
  const int lrow  = mbase + l16;
  const int crow  = lrow < NNODES ? lrow : NNODES - 1;   // clamped load row
  const float scale = rOut[crow];
  const float* xrow = X + (size_t)crow * FIN;

  v8f acc[4] = {};
  for (int ks = 0; ks < FIN; ks += 32) {
    // A fragment (16-bit A layout): lanes 0-15 hold K=ks+{0..7,16..23},
    // lanes 16-31 hold K=ks+{8..15,24..31}, row M = lane&15.
    const float* rp = xrow + ks + half * 8;
    const float4* p0 = reinterpret_cast<const float4*>(rp);
    const float4* p1 = reinterpret_cast<const float4*>(rp + 16);
    float4 a0 = p0[0], a1 = p0[1], c0 = p1[0], c1 = p1[1];
    const float t[16] = {a0.x,a0.y,a0.z,a0.w, a1.x,a1.y,a1.z,a1.w,
                         c0.x,c0.y,c0.z,c0.w, c1.x,c1.y,c1.z,c1.w};
    bf16v afrag;
#pragma unroll
    for (int i = 0; i < 16; ++i) afrag[i] = (__bf16)(t[i] * scale);

#pragma unroll
    for (int nt = 0; nt < 4; ++nt) {
      const int col = nt * 16 + l16;
      const uint4* bp = reinterpret_cast<const uint4*>(&sW[col * 264 + ks + half * 8]);
      bf16v bfrag = cat8(bp[0], bp[2]);
      acc[nt] = __builtin_amdgcn_wmma_f32_16x16x32_bf16(
          false, afrag, false, bfrag, (short)0, acc[nt], false, false);
    }
  }
  if (mbase < NNODES) {                 // tile fully in range (never partial)
    // C layout: element i of lane -> row mbase + half*8 + i, col nt*16 + l16
    float* op = H1 + (size_t)(mbase + half * 8) * FHID + l16;
#pragma unroll
    for (int nt = 0; nt < 4; ++nt)
#pragma unroll
      for (int i = 0; i < 8; ++i)
        op[i * FHID + nt * 16] = acc[nt][i];   // constant immediate offsets
  }
}

// ----------------------- edge scatter (segment_sum) ------------------------
__global__ void gcn_scatter1(const int* __restrict__ src, const int* __restrict__ dst,
                             const float* __restrict__ H1, float* __restrict__ agg) {
  const int g = blockIdx.x * blockDim.x + threadIdx.x;   // exactly E*64 threads
  const int e = g >> 6, f = g & 63;
  atomicAdd(&agg[(size_t)dst[e] * FHID + f], H1[(size_t)src[e] * FHID + f]);
}

// -------------------- exact jax threefry2x32 (key = 42) --------------------
__device__ __forceinline__ unsigned rotl32(unsigned x, int r) {
  return (x << r) | (x >> (32 - r));
}
__device__ __forceinline__ unsigned threefry_word(unsigned idx) {
  const unsigned halfN = (unsigned)(NNODES * FHID) / 2u;
  unsigned x0, x1; bool second;
  if (idx < halfN) { x0 = idx;         x1 = idx + halfN; second = false; }
  else             { x0 = idx - halfN; x1 = idx;         second = true;  }
  const unsigned k0 = 0u, k1 = 42u, k2 = k0 ^ k1 ^ 0x1BD11BDAu;
  const int ra[4] = {13, 15, 26, 6}, rb[4] = {17, 29, 16, 24};
  x0 += k0; x1 += k1;
#pragma unroll
  for (int i = 0; i < 4; ++i) { x0 += x1; x1 = rotl32(x1, ra[i]); x1 ^= x0; }
  x0 += k1; x1 += k2 + 1u;
#pragma unroll
  for (int i = 0; i < 4; ++i) { x0 += x1; x1 = rotl32(x1, rb[i]); x1 ^= x0; }
  x0 += k2; x1 += k0 + 2u;
#pragma unroll
  for (int i = 0; i < 4; ++i) { x0 += x1; x1 = rotl32(x1, ra[i]); x1 ^= x0; }
  x0 += k0; x1 += k1 + 3u;
#pragma unroll
  for (int i = 0; i < 4; ++i) { x0 += x1; x1 = rotl32(x1, rb[i]); x1 ^= x0; }
  x0 += k1; x1 += k2 + 4u;
#pragma unroll
  for (int i = 0; i < 4; ++i) { x0 += x1; x1 = rotl32(x1, ra[i]); x1 ^= x0; }
  x0 += k2; x1 += k0 + 5u;
  return second ? x1 : x0;
}

// bias + relu + dropout(0.5, exact mask) + rsqrt(deg_out) prescale -> bf16
__global__ void gcn_postact(const float* __restrict__ agg, const float* __restrict__ rIn,
                            const float* __restrict__ rOut, const float* __restrict__ b1,
                            unsigned short* __restrict__ hs) {
  const int idx = blockIdx.x * blockDim.x + threadIdx.x;  // exactly N*64 threads
  const int i = idx >> 6, f = idx & 63;
  float v = agg[idx] * rIn[i] + b1[f];
  v = fmaxf(v, 0.0f);
  // uniform < 0.5  <=>  MSB of random word == 0 ; scale kept values by 1/(1-p)=2
  const bool keep = (threefry_word((unsigned)idx) & 0x80000000u) == 0u;
  hs[idx] = f2bf((keep ? v * 2.0f : 0.0f) * rOut[i]);
}

// ----------------------- GEMM2: H2 = HS(bf16) @ W2 -------------------------
__global__ void __launch_bounds__(256)
gcn_gemm2(const unsigned short* __restrict__ HS, const float* __restrict__ W2,
          float* __restrict__ H2) {
  __shared__ __align__(16) unsigned short sW[NCP * 72];   // W2^T padded 40->48 cols
  const int tid = threadIdx.x;
  for (int idx = tid; idx < NCP * FHID; idx += 256) {
    const int c = idx >> 6, k = idx & 63;                 // W2 is [64,40] row-major
    sW[c * 72 + k] = (c < NCLS) ? f2bf(W2[k * NCLS + c]) : (unsigned short)0;
  }
  __syncthreads();

  const int wave = tid >> 5;
  const int lane = tid & 31;
  const int half = lane >> 4;
  const int l16  = lane & 15;
  const int mbase = (blockIdx.x * 8 + wave) * 16;
  const int lrow  = mbase + l16;
  const int crow  = lrow < NNODES ? lrow : NNODES - 1;    // clamped load row
  const unsigned short* hrow = HS + (size_t)crow * FHID;

  v8f acc[3] = {};
#pragma unroll
  for (int ks = 0; ks < FHID; ks += 32) {
    const unsigned short* rp = hrow + ks + half * 8;
    uint4 lo = *reinterpret_cast<const uint4*>(rp);
    uint4 hi = *reinterpret_cast<const uint4*>(rp + 16);
    bf16v afrag = cat8(lo, hi);
#pragma unroll
    for (int nt = 0; nt < 3; ++nt) {
      const int col = nt * 16 + l16;
      const uint4* bp = reinterpret_cast<const uint4*>(&sW[col * 72 + ks + half * 8]);
      bf16v bfrag = cat8(bp[0], bp[2]);
      acc[nt] = __builtin_amdgcn_wmma_f32_16x16x32_bf16(
          false, afrag, false, bfrag, (short)0, acc[nt], false, false);
    }
  }
  if (mbase < NNODES) {                 // tile fully in range (never partial)
    float* op = H2 + (size_t)(mbase + half * 8) * NCP + l16;
#pragma unroll
    for (int nt = 0; nt < 3; ++nt)
#pragma unroll
      for (int i = 0; i < 8; ++i)
        op[i * NCP + nt * 16] = acc[nt][i];
  }
}

__global__ void gcn_scatter2(const int* __restrict__ src, const int* __restrict__ dst,
                             const float* __restrict__ H2, float* __restrict__ out) {
  const int g = blockIdx.x * blockDim.x + threadIdx.x;   // exactly E*40 threads
  const int e = g / NCLS, f = g - e * NCLS;
  atomicAdd(&out[(size_t)dst[e] * NCLS + f], H2[(size_t)src[e] * NCP + f]);
}

__global__ void gcn_final(float* __restrict__ out, const float* __restrict__ rIn,
                          const float* __restrict__ b2) {
  const int g = blockIdx.x * blockDim.x + threadIdx.x;   // exactly N*40 threads
  const int i = g / NCLS, f = g - i * NCLS;
  out[g] = out[g] * rIn[i] + b2[f];
}

// --------------------------------- launch ----------------------------------
extern "C" void kernel_launch(void* const* d_in, const int* in_sizes, int n_in,
                              void* d_out, int out_size, void* d_ws, size_t ws_size,
                              hipStream_t stream) {
  (void)in_sizes; (void)n_in; (void)out_size; (void)ws_size;
  const float* X  = (const float*)d_in[0];
  const float* W1 = (const float*)d_in[1];
  const float* b1 = (const float*)d_in[2];
  const float* W2 = (const float*)d_in[3];
  const float* b2 = (const float*)d_in[4];
  const int*   src = (const int*)d_in[5];
  const int*   dst = (const int*)d_in[6];
  float* out = (float*)d_out;

  // workspace layout (ws re-initialized every call; deterministic)
  float* degOut = (float*)d_ws;                       // [N]   -> rOut (in place)
  float* degIn  = degOut + NNODES;                    // [N]   -> rIn  (in place)
  float* h1     = degIn + NNODES;                     // [N*64] f32, reused as H2 (stride 48)
  float* agg1   = h1 + (size_t)NNODES * FHID;         // [N*64] f32
  unsigned short* hs = (unsigned short*)(agg1 + (size_t)NNODES * FHID); // [N*64] bf16
  float* h2 = h1;                                     // h1 dead before GEMM2 writes

  gcn_zero<<<(2 * NNODES + 255) / 256, 256, 0, stream>>>(degOut, 2 * NNODES);
  gcn_zero<<<(NNODES * FHID + 255) / 256, 256, 0, stream>>>(agg1, NNODES * FHID);
  gcn_zero<<<(NNODES * NCLS + 255) / 256, 256, 0, stream>>>(out, NNODES * NCLS);

  gcn_degrees<<<NEDGES / 256, 256, 0, stream>>>(src, dst, degOut, degIn);
  gcn_rsqrt_inplace<<<(2 * NNODES + 255) / 256, 256, 0, stream>>>(degOut, 2 * NNODES);

  const int mblocks = (NNODES + 127) / 128;           // 8 waves x 16 rows per block
  gcn_gemm1<<<mblocks, 256, 0, stream>>>(X, W1, degOut, h1);
  gcn_scatter1<<<NEDGES * 64 / 256, 256, 0, stream>>>(src, dst, h1, agg1);
  gcn_postact<<<NNODES * FHID / 256, 256, 0, stream>>>(agg1, degIn, degOut, b1, hs);
  gcn_gemm2<<<mblocks, 256, 0, stream>>>(hs, W2, h2);
  gcn_scatter2<<<NEDGES * NCLS / 256, 256, 0, stream>>>(src, dst, h2, out);
  gcn_final<<<NNODES * NCLS / 256, 256, 0, stream>>>(out, degIn, b2);
}
